// ProcessingCov_HyperLeaf2024_SCM_15444702397096
// MI455X (gfx1250) — matrix operand order repaired and checked by reading, not compiled, not run on hardware
//
#include <hip/hip_runtime.h>
#include <hip/hip_bf16.h>

// ---------------------------------------------------------------------------
// SCM (per-image covariance, all-zero pixels excluded) for MI455X gfx1250.
//   cov[b] = Gram[b]/n[b] - mean[b] mean[b]^T
// Zero pixels contribute nothing to Gram or sums (they are exactly 0); the
// mask only affects the count n. Gram via split-bf16 (hi+lo):
//   x ~= hi + lo  =>  x*y ~= hi*hi + hi*lo + lo*hi   (3x wmma bf16, f32 acc)
// hi/lo planes are produced ONCE during staging and stored as bf16 in LDS;
// all fragment loads are ds_load_b128; only upper-triangle tiles computed.
// Tile 0..3 run guard-free on every wave; only the 5th tile (waves 0..3)
// needs a branch, made scalar via readfirstlane so EXEC is never masked.
// ---------------------------------------------------------------------------

typedef __bf16 bf16x16 __attribute__((ext_vector_type(16)));
typedef __bf16 bf16x8  __attribute__((ext_vector_type(8)));
typedef float  f32x8   __attribute__((ext_vector_type(8)));

#define NB              32
#define NPIX            65536      // 256*256
#define NCH             128
#define THREADS         256        // 8 waves (wave32)
#define PIXCHUNK        32         // pixels staged per iteration (== WMMA K)
#define CHUNKS          64         // pixels per block = 2048
#define BLOCKS_PER_IMG  32         // 32*2048 = 65536
#define PITCH           40         // bf16 elements per LDS row (80B, 16B mult.)
#define NTILES          36         // upper triangle of 8x8 tile grid

// Split f32 into bf16 hi + bf16 lo (round-half-up on the 16-bit boundary).
__device__ __forceinline__ void split_bf16(float f, __bf16& h, __bf16& l) {
  unsigned u  = __float_as_uint(f);
  unsigned uh = (u + 0x8000u) & 0xffff0000u;
  h = __builtin_bit_cast(__bf16, (unsigned short)(uh >> 16));
  float r = f - __uint_as_float(uh);              // exact residual
  unsigned ur = __float_as_uint(r);
  l = __builtin_bit_cast(__bf16, (unsigned short)((ur + 0x8000u) >> 16));
}

__device__ __forceinline__ bf16x16 cat8(bf16x8 a, bf16x8 b) {
  return __builtin_shufflevector(a, b, 0, 1, 2, 3, 4, 5, 6, 7,
                                       8, 9, 10, 11, 12, 13, 14, 15);
}

// A fragment (16x32 bf16): elements 0..7 <-> K = 8*hl + e,
//                          elements 8..15 <-> K = 16 + 8*hl + (e-8).
__device__ __forceinline__ bf16x16 load_fragA(const __bf16* plane, int ch, int hl) {
  const bf16x8 a0 = *reinterpret_cast<const bf16x8*>(plane + ch * PITCH + 8 * hl);
  const bf16x8 a1 = *reinterpret_cast<const bf16x8*>(plane + ch * PITCH + 16 + 8 * hl);
  return cat8(a0, a1);
}
// B fragment (32x16 bf16): elements 0..7 <-> K = 8*hl + e,
//                          elements 8..15 <-> K = 16 + 8*hl + (e-8).
__device__ __forceinline__ bf16x16 load_fragB(const __bf16* plane, int ch, int hl) {
  const bf16x8 b0 = *reinterpret_cast<const bf16x8*>(plane + ch * PITCH + 8 * hl);
  const bf16x8 b1 = *reinterpret_cast<const bf16x8*>(plane + ch * PITCH + 16 + 8 * hl);
  return cat8(b0, b1);
}

__global__ void scm_zero(float* ws, int n) {
  int i = blockIdx.x * 256 + threadIdx.x;
  if (i < n) ws[i] = 0.0f;
}

__global__ __launch_bounds__(THREADS)
void scm_gram(const float* __restrict__ X,
              float* __restrict__ G,      // [NB][NCH][NCH] partial Gram (atomic)
              float* __restrict__ S,      // [NB][NCH] channel sums (atomic)
              float* __restrict__ CNT) {  // [NB] nonzero-pixel counts (atomic)
  __shared__ __bf16 hiP[NCH * PITCH];     // transposed bf16 hi plane [ch][pix]
  __shared__ __bf16 loP[NCH * PITCH];     // transposed bf16 lo plane [ch][pix]
  __shared__ float ssum[NCH];
  __shared__ unsigned nzmask;             // bit p = pixel p (of chunk) nonzero

  const int tid  = threadIdx.x;
  const int wave = tid >> 5;
  const int lane = tid & 31;
  const int b    = blockIdx.x / BLOCKS_PER_IMG;
  const int blk  = blockIdx.x % BLOCKS_PER_IMG;
  const long p0  = (long)blk * (PIXCHUNK * CHUNKS);
  const float* Ximg = X + (long)b * NPIX * NCH;

  // staging mapping: this thread owns channel c, pixel groups of 8
  const int c = tid & 127;                // owned channel
  const int g = tid >> 7;                 // pixel half: 0 -> 0..15, 1 -> 16..31
  const int m  = lane & 15;               // row-in-tile (A) / col-in-tile (B,D)
  const int hl = lane >> 4;               // lane half selects K sub-range

  // scalar (SGPR) copy of the wave id -> guards become s_cmp/s_cbranch,
  // EXEC stays all-ones around every WMMA.
  const int swave = __builtin_amdgcn_readfirstlane(wave);
  const bool has5 = (swave < (NTILES - 32));         // waves 0..3 own a 5th tile

  // balanced upper-triangle tile assignment: wave w owns tiles w, w+8, ...
  int ti[5], tj[5];
  #pragma unroll
  for (int t = 0; t < 5; ++t) {
    int idx = wave + 8 * t;
    if (idx >= NTILES) idx = NTILES - 1;             // clamp (unused slot)
    int ii = 0, r = idx;
    while (r >= 8 - ii) { r -= 8 - ii; ++ii; }       // decode triangle index
    ti[t] = ii; tj[t] = ii + r;
  }

  if (tid < NCH) ssum[tid] = 0.0f;
  if (tid == 0) nzmask = 0u;

  f32x8 acc[5];
  #pragma unroll
  for (int t = 0; t < 5; ++t) acc[t] = (f32x8){0, 0, 0, 0, 0, 0, 0, 0};

  float sacc = 0.0f;                      // per-channel running sum
  int cnt = 0;                            // valid pixels (held by tid 0)

  for (int ch = 0; ch < CHUNKS; ++ch) {
    __syncthreads();
    // ---- stage: 32 pixels x 128 channels -> bf16 hi/lo planes -------------
    unsigned pm = 0u;                     // local nonzero mask (16 pixels)
    #pragma unroll
    for (int q = 0; q < 2; ++q) {
      const int pb = g * 16 + q * 8;      // 8 consecutive pixels
      const float* src = Ximg + (p0 + (long)ch * PIXCHUNK + pb) * NCH + c;
      bf16x8 hv, lv;
      #pragma unroll
      for (int k = 0; k < 8; ++k) {
        const float f = __builtin_nontemporal_load(src + (long)k * NCH);
        sacc += f;
        if (f != 0.0f) pm |= 1u << (pb + k);
        __bf16 h, l; split_bf16(f, h, l);
        hv[k] = h; lv[k] = l;
      }
      *reinterpret_cast<bf16x8*>(hiP + c * PITCH + pb) = hv;
      *reinterpret_cast<bf16x8*>(loP + c * PITCH + pb) = lv;
      if (ch + 1 < CHUNKS)                // gfx1250 global_prefetch_b8
        __builtin_prefetch(src + PIXCHUNK * NCH, 0, 0);
    }
    // wave-level OR reduce of the pixel mask, one LDS atomic per wave
    #pragma unroll
    for (int off = 16; off >= 1; off >>= 1) pm |= __shfl_xor(pm, off);
    if (lane == 0) atomicOr(&nzmask, pm);
    __syncthreads();

    if (tid == 0) { cnt += __popc(nzmask); nzmask = 0u; }

    // ---- upper-triangle tiles: ds_load_b128 fragments + 3 WMMAs each ------
    #pragma unroll
    for (int t = 0; t < 4; ++t) {         // every wave owns >= 4 tiles
      const int ca = 16 * ti[t] + m;      // A channel row
      const int cb = 16 * tj[t] + m;      // B channel col
      const bf16x16 a_hi = load_fragA(hiP, ca, hl);
      const bf16x16 a_lo = load_fragA(loP, ca, hl);
      const bf16x16 b_hi = load_fragB(hiP, cb, hl);
      const bf16x16 b_lo = load_fragB(loP, cb, hl);
      acc[t] = __builtin_amdgcn_wmma_f32_16x16x32_bf16(
                   false, a_hi, false, b_hi, (short)0, acc[t], false, false);
      acc[t] = __builtin_amdgcn_wmma_f32_16x16x32_bf16(
                   false, a_hi, false, b_lo, (short)0, acc[t], false, false);
      acc[t] = __builtin_amdgcn_wmma_f32_16x16x32_bf16(
                   false, a_lo, false, b_hi, (short)0, acc[t], false, false);
    }
    if (has5) {                           // scalar branch, EXEC untouched
      const int ca = 16 * ti[4] + m;
      const int cb = 16 * tj[4] + m;
      const bf16x16 a_hi = load_fragA(hiP, ca, hl);
      const bf16x16 a_lo = load_fragA(loP, ca, hl);
      const bf16x16 b_hi = load_fragB(hiP, cb, hl);
      const bf16x16 b_lo = load_fragB(loP, cb, hl);
      acc[4] = __builtin_amdgcn_wmma_f32_16x16x32_bf16(
                   false, a_hi, false, b_hi, (short)0, acc[4], false, false);
      acc[4] = __builtin_amdgcn_wmma_f32_16x16x32_bf16(
                   false, a_hi, false, b_lo, (short)0, acc[4], false, false);
      acc[4] = __builtin_amdgcn_wmma_f32_16x16x32_bf16(
                   false, a_lo, false, b_hi, (short)0, acc[4], false, false);
    }
  }

  // ---- reduce channel sums via LDS, then one global atomic per channel ----
  __syncthreads();
  atomicAdd(&ssum[c], sacc);
  __syncthreads();
  if (tid < NCH) atomicAdd(&S[b * NCH + tid], ssum[tid]);
  if (tid == 0) atomicAdd(&CNT[b], (float)cnt);

  // ---- accumulate partial Gram (C/D 16x16 f32 layout: M = vv + 8*hl) ------
  #pragma unroll
  for (int t = 0; t < 4; ++t) {
    const int crow = 16 * ti[t] + 8 * hl;
    const int d    = 16 * tj[t] + m;
    #pragma unroll
    for (int vv = 0; vv < 8; ++vv) {
      atomicAdd(&G[((long)b * NCH + (crow + vv)) * NCH + d], acc[t][vv]);
    }
  }
  if (has5) {
    const int crow = 16 * ti[4] + 8 * hl;
    const int d    = 16 * tj[4] + m;
    #pragma unroll
    for (int vv = 0; vv < 8; ++vv) {
      atomicAdd(&G[((long)b * NCH + (crow + vv)) * NCH + d], acc[4][vv]);
    }
  }
}

__global__ void scm_finalize(const float* __restrict__ G,
                             const float* __restrict__ S,
                             const float* __restrict__ CNT,
                             float* __restrict__ out) {
  int idx = blockIdx.x * 256 + threadIdx.x;
  if (idx >= NB * NCH * NCH) return;
  int b = idx >> 14;                      // NCH*NCH = 16384
  int rem = idx & (NCH * NCH - 1);
  int cc = rem >> 7;
  int dd = rem & (NCH - 1);
  // only upper-triangle tiles were accumulated; mirror for the lower ones
  int c = cc, d = dd;
  if ((cc >> 4) > (dd >> 4)) { c = dd; d = cc; }
  float inv = 1.0f / CNT[b];
  out[idx] = G[((long)b * NCH + c) * NCH + d] * inv
           - (S[b * NCH + cc] * inv) * (S[b * NCH + dd] * inv);
}

extern "C" void kernel_launch(void* const* d_in, const int* in_sizes, int n_in,
                              void* d_out, int out_size, void* d_ws, size_t ws_size,
                              hipStream_t stream) {
  (void)in_sizes; (void)n_in; (void)out_size; (void)ws_size;
  const float* X = (const float*)d_in[0];
  float* ws  = (float*)d_ws;
  float* G   = ws;                                  // NB*NCH*NCH floats (2 MB)
  float* S   = G + (long)NB * NCH * NCH;            // NB*NCH floats
  float* CNT = S + NB * NCH;                        // NB floats

  const int zn = NB * NCH * NCH + NB * NCH + NB;
  scm_zero<<<(zn + 255) / 256, 256, 0, stream>>>(ws, zn);
  scm_gram<<<NB * BLOCKS_PER_IMG, THREADS, 0, stream>>>(X, G, S, CNT);
  scm_finalize<<<(NB * NCH * NCH + 255) / 256, 256, 0, stream>>>(G, S, CNT,
                                                                 (float*)d_out);
}
